// SelfAttnV2EOpt_10290741641924
// MI455X (gfx1250) — matrix operand order repaired and checked by reading, not compiled, not run on hardware
//
#include <hip/hip_runtime.h>

// ---------------------------------------------------------------- constants
#define NN      100000
#define EE      50000
#define NNZ     600000
#define DD      256
#define MAXL1   11          // MAXL+1 orders
#define NEGS    0.2f
#define INV_SQRT_DQH 0.17677669529663687f   // 1/sqrt(32)

#define TILE_M  64          // rows per workgroup in GEMM kernels
#define GEMM_THREADS 128    // 4 waves * 32 lanes, each wave owns 16 rows
#define WBYTES  131072      // one packed 256x256 bf16 weight = 128KB

// ---------------------------------------------------------------- types
typedef __attribute__((ext_vector_type(16))) __bf16          v16bf;
typedef __attribute__((ext_vector_type(8)))  float           v8f;
typedef __attribute__((ext_vector_type(8)))  unsigned short  u16x8;
typedef __attribute__((ext_vector_type(16))) unsigned short  u16x16;

// f32 -> bf16 (round to nearest even), bit-level (safe on all toolchains)
__device__ __forceinline__ unsigned short f2bf(float f) {
    unsigned u = __builtin_bit_cast(unsigned, f);
    unsigned r = u + 0x7FFFu + ((u >> 16) & 1u);
    return (unsigned short)(r >> 16);
}

// order-preserving float <-> uint encoding for atomicMax on floats
__device__ __forceinline__ unsigned fenc(float f) {
    unsigned u = __builtin_bit_cast(unsigned, f);
    return (u & 0x80000000u) ? ~u : (u | 0x80000000u);
}
__device__ __forceinline__ float fdec(unsigned u) {
    unsigned b = (u & 0x80000000u) ? (u & 0x7FFFFFFFu) : ~u;
    return __builtin_bit_cast(float, b);
}

__device__ __forceinline__ v8f wmma_bf16(v16bf a, v16bf b, v8f c) {
    return __builtin_amdgcn_wmma_f32_16x16x32_bf16(false, a, false, b,
                                                   (short)0, c, false, false);
}

// CDNA5 async global->LDS copy (ASYNCcnt-tracked, no VGPR round trip).
// lds_addr = low 32 bits of the flat LDS address (= LDS byte offset per ISA
// aperture rules).
__device__ __forceinline__ void async_g2l_b128(unsigned lds_addr, const void* gaddr) {
    asm volatile("global_load_async_to_lds_b128 %0, %1, off"
                 :: "v"(lds_addr), "v"(gaddr) : "memory");
}
__device__ __forceinline__ void wait_async0() {
    asm volatile("s_wait_asynccnt 0" ::: "memory");
}
__device__ __forceinline__ unsigned lds_lo32(const void* p) {
    return (unsigned)(unsigned long long)p;
}

// A-fragment (16x32 bf16, M=lane&15; lanes<16 hold K {k0..k0+7, k0+16..k0+23},
// lanes>=16 hold K {k0+8..k0+15, k0+24..k0+31}); p already lane-adjusted.
__device__ __forceinline__ v16bf load_a_frag(const unsigned short* p) {
    u16x8 lo = *(const u16x8*)p;          // 16B ds_load
    u16x8 hi = *(const u16x8*)(p + 16);   // 16B ds_load
    u16x16 t;
#pragma unroll
    for (int i = 0; i < 8; ++i) { t[i] = lo[i]; t[8 + i] = hi[i]; }
    return __builtin_bit_cast(v16bf, t);
}

// B-fragment: packed contiguous 16 halfs per lane (32B, from LDS-staged weights)
__device__ __forceinline__ v16bf load_b_frag(const unsigned short* p) {
    u16x16 t = *(const u16x16*)p;
    return __builtin_bit_cast(v16bf, t);
}

// ---------------------------------------------------------------- weight packing
// Pack a 256x256 f32 weight W[k][n] (row stride 256) into WMMA-B fragment order:
// dst[((kt*16+nt)*32 + lane)*16 + e], K=kt*32 + 8*(lane>>4) + (e&7) + 16*(e>>3),
// N = nt*16 + (lane&15).
__global__ void pack_weight_kernel(const float* __restrict__ W,
                                   unsigned short* __restrict__ Wp) {
    int i = blockIdx.x * 256 + threadIdx.x;      // 0 .. 65535
    int e  = i & 15;
    int l  = (i >> 4) & 31;
    int nt = (i >> 9) & 15;
    int kt = (i >> 13) & 7;
    int n = nt * 16 + (l & 15);
    int k = kt * 32 + ((l >> 4) << 3) + (e & 7) + ((e >> 3) << 4);
    Wp[i] = f2bf(W[k * 256 + n]);
}

// ---------------------------------------------------------------- small prep
__global__ __launch_bounds__(256)
void small_prep_kernel(const float* pe_q, const float* qw1, const float* qb1,
                       const float* qw2, const float* qb2,
                       const float* kw, const float* kb,
                       const float* pe_b, const float* bw1, const float* bb1,
                       const float* bw2, const float* bb2,
                       const float* pe2, const float* m2w1, const float* m2b1,
                       const float* pe3, const float* m3w1, const float* m3b1,
                       float* wl0, float* wl1, float* bl0, float* bl1,
                       float* bias_out, float* b2eff, float* c3) {
    __shared__ float hq[2 * 256];
    __shared__ float q01s[2 * 256];
    __shared__ float hb[MAXL1 * 256];
    const int j = threadIdx.x;

    for (int p = 0; p < 2; ++p) {
        float a = qb1[j];
        for (int c = 0; c < 64; ++c) a += pe_q[p * 64 + c] * qw1[c * 256 + j];
        hq[p * 256 + j] = fmaxf(a, 0.f);
    }
    __syncthreads();
    for (int p = 0; p < 2; ++p) {
        float a = qb2[j];
        for (int c = 0; c < 256; ++c) a += hq[p * 256 + c] * qw2[c * 256 + j];
        q01s[p * 256 + j] = a;
    }
    __syncthreads();

    // fold kw with q0/q1:  wl0[h*256+d] = sum_c kw[d][h*32+c]*q0[h][c]
    for (int idx = j; idx < 8 * 256; idx += 256) {
        int h = idx >> 8, d = idx & 255;
        float a0 = 0.f, a1 = 0.f;
        for (int c = 0; c < 32; ++c) {
            a0 += kw[d * 512 + h * 32 + c]       * q01s[h * 32 + c];
            a1 += kw[d * 512 + 256 + h * 32 + c] * q01s[256 + h * 32 + c];
        }
        wl0[h * 256 + d] = a0;
        wl1[h * 256 + d] = a1;
    }
    if (j < 8) {
        float a0 = 0.f, a1 = 0.f;
        for (int c = 0; c < 32; ++c) {
            a0 += kb[j * 32 + c]       * q01s[j * 32 + c];
            a1 += kb[256 + j * 32 + c] * q01s[256 + j * 32 + c];
        }
        bl0[j] = a0; bl1[j] = a1;
    }

    for (int o = 0; o < MAXL1; ++o) {
        float a = bb1[j];
        for (int c = 0; c < 64; ++c) a += pe_b[o * 64 + c] * bw1[c * 256 + j];
        hb[o * 256 + j] = fmaxf(a, 0.f);
    }
    __syncthreads();
    for (int o = 0; o < MAXL1; ++o) {
        float a = bb2[j];
        for (int c = 0; c < 256; ++c) a += hb[o * 256 + c] * bw2[c * 256 + j];
        bias_out[o * 256 + j] = a;
    }

    for (int t = 0; t < 2; ++t) {
        float a = m2b1[j];
        for (int d = 0; d < 256; ++d) a += pe2[t * 256 + d] * m2w1[(256 + d) * 256 + j];
        b2eff[t * 256 + j] = a;
    }
    for (int o = 0; o < MAXL1; ++o) {
        float a = m3b1[j];
        for (int d = 0; d < 256; ++d) a += pe3[o * 256 + d] * m3w1[(256 + d) * 256 + j];
        c3[o * 256 + j] = a;
    }
}

// ---------------------------------------------------------------- fused residual block
// out = Tin + relu(LN(Tin)@W1 + b1_eff[order(row)]) @ W2 + b2  (+extra) (+post_bias[order])
// 4 waves / 64 rows per WG. Weights async-staged to LDS once per block and
// shared by all waves (W1, then W2 in the same 128KB region).
__global__ __launch_bounds__(GEMM_THREADS)
void fused_block_kernel(const float* __restrict__ Tin, float* __restrict__ Tout,
                        int rows, const float* __restrict__ lng, const float* __restrict__ lnb,
                        const unsigned short* __restrict__ W1p,
                        const float* __restrict__ b1_base, const int* __restrict__ ord_idx,
                        int ord_fixed,
                        const unsigned short* __restrict__ W2p, const float* __restrict__ b2,
                        const float* __restrict__ extra,
                        const float* __restrict__ post_base, int post_fixed) {
    extern __shared__ unsigned char smem[];
    unsigned short* sA = (unsigned short*)smem;                        // 64x256 bf16 (32KB)
    unsigned short* sH = (unsigned short*)(smem + TILE_M * DD * 2);    // 64x256 bf16 (32KB)
    unsigned short* sW = (unsigned short*)(smem + 2 * TILE_M * DD * 2);// 128KB weight stage
    __shared__ int sOrd[TILE_M];

    const int tid  = threadIdx.x;
    const int wave = tid >> 5;
    const int lane = tid & 31;
    const int l15  = lane & 15;
    const int lhalf = lane >> 4;
    const int r0 = wave * 16;
    const long long tile = (long long)blockIdx.x * TILE_M;

    // ---- kick off async staging of W1 into LDS (overlaps with LN below)
    {
        const unsigned base_lds = lds_lo32(sW);
#pragma unroll 4
        for (int i = tid; i < WBYTES / 16; i += GEMM_THREADS)
            async_g2l_b128(base_lds + i * 16, (const char*)W1p + i * 16);
    }

    if (ord_idx) {
        for (int i = tid; i < TILE_M; i += GEMM_THREADS) {
            long long rg = tile + i; if (rg >= rows) rg = rows - 1;
            sOrd[i] = ord_idx[rg];
        }
    }

    // ---- stage 1: LayerNorm rows -> bf16 in sA (wave-local)
    const int c0 = lane * 8;
    float4 g0 = *(const float4*)(lng + c0);
    float4 g1 = *(const float4*)(lng + c0 + 4);
    float4 bt0 = *(const float4*)(lnb + c0);
    float4 bt1 = *(const float4*)(lnb + c0 + 4);
    float gs[8] = {g0.x, g0.y, g0.z, g0.w, g1.x, g1.y, g1.z, g1.w};
    float bs[8] = {bt0.x, bt0.y, bt0.z, bt0.w, bt1.x, bt1.y, bt1.z, bt1.w};

    for (int rr = 0; rr < 16; ++rr) {
        const int rl = r0 + rr;
        long long rg = tile + rl; if (rg >= rows) rg = rows - 1;
        const float* rp = Tin + rg * DD + c0;
        float4 a0 = *(const float4*)rp;
        float4 a1 = *(const float4*)(rp + 4);
        float vals[8] = {a0.x, a0.y, a0.z, a0.w, a1.x, a1.y, a1.z, a1.w};
        float s = 0.f, q = 0.f;
#pragma unroll
        for (int i = 0; i < 8; ++i) { s += vals[i]; q += vals[i] * vals[i]; }
#pragma unroll
        for (int off = 16; off >= 1; off >>= 1) {
            s += __shfl_xor(s, off, 32);
            q += __shfl_xor(q, off, 32);
        }
        const float mean = s * (1.f / 256.f);
        const float var  = q * (1.f / 256.f) - mean * mean;
        const float rstd = rsqrtf(var + 1e-5f);
        u16x8 o;
#pragma unroll
        for (int i = 0; i < 8; ++i)
            o[i] = f2bf((vals[i] - mean) * rstd * gs[i] + bs[i]);
        *(u16x8*)(sA + rl * DD + c0) = o;
    }

    wait_async0();       // W1 resident (this wave's share)
    __syncthreads();     // all waves' shares visible

    // ---- GEMM1: hidden = relu(LN(T)@W1 + b1_eff) -> sH (bf16)
    v16bf aF[8];
#pragma unroll
    for (int kt = 0; kt < 8; ++kt)
        aF[kt] = load_a_frag(sA + (r0 + l15) * DD + kt * 32 + lhalf * 8);

    for (int nt = 0; nt < 16; ++nt) {
        v8f acc = {0.f, 0.f, 0.f, 0.f, 0.f, 0.f, 0.f, 0.f};
#pragma unroll
        for (int kt = 0; kt < 8; ++kt) {
            v16bf bF = load_b_frag(sW + (((kt * 16 + nt) * 32) + lane) * 16);
            acc = wmma_bf16(aF[kt], bF, acc);
        }
        const int col = nt * 16 + l15;
        const float bfix = (!ord_idx) ? b1_base[ord_fixed * DD + col] : 0.f;
#pragma unroll
        for (int r = 0; r < 8; ++r) {
            const int rl = r0 + lhalf * 8 + r;
            const float bv = ord_idx ? b1_base[sOrd[rl] * DD + col] : bfix;
            float h = acc[r] + bv;
            h = h > 0.f ? h : 0.f;
            sH[rl * DD + col] = f2bf(h);
        }
    }

    __syncthreads();     // all waves done reading W1 from sW

    // ---- async restage W2 into the same LDS region
    {
        const unsigned base_lds = lds_lo32(sW);
#pragma unroll 4
        for (int i = tid; i < WBYTES / 16; i += GEMM_THREADS)
            async_g2l_b128(base_lds + i * 16, (const char*)W2p + i * 16);
    }

    // preload GEMM2 A-fragments from sH while W2 is in flight
#pragma unroll
    for (int kt = 0; kt < 8; ++kt)
        aF[kt] = load_a_frag(sH + (r0 + l15) * DD + kt * 32 + lhalf * 8);

    wait_async0();
    __syncthreads();

    // ---- GEMM2 + epilogue
    for (int nt = 0; nt < 16; ++nt) {
        v8f acc = {0.f, 0.f, 0.f, 0.f, 0.f, 0.f, 0.f, 0.f};
#pragma unroll
        for (int kt = 0; kt < 8; ++kt) {
            v16bf bF = load_b_frag(sW + (((kt * 16 + nt) * 32) + lane) * 16);
            acc = wmma_bf16(aF[kt], bF, acc);
        }
        const int col = nt * 16 + l15;
        float badd = b2[col];
        if (extra) badd += extra[col];
        const float pfix = (post_base && !ord_idx) ? post_base[post_fixed * DD + col] : 0.f;
#pragma unroll
        for (int r = 0; r < 8; ++r) {
            const int rl = r0 + lhalf * 8 + r;
            const long long rg = tile + rl;
            if (rg < rows) {
                float res = Tin[rg * DD + col];
                float pv = post_base ? (ord_idx ? post_base[sOrd[rl] * DD + col] : pfix) : 0.f;
                Tout[rg * DD + col] = acc[r] + badd + res + pv;
            }
        }
    }
}

// ---------------------------------------------------------------- single GEMM + bias (v = x1@vw + vb)
__global__ __launch_bounds__(GEMM_THREADS)
void gemm_bias_kernel(const float* __restrict__ Tin, float* __restrict__ Tout,
                      int rows, const unsigned short* __restrict__ Wp,
                      const float* __restrict__ bias) {
    extern __shared__ unsigned char smem[];
    unsigned short* sA = (unsigned short*)smem;                      // 32KB
    unsigned short* sW = (unsigned short*)(smem + TILE_M * DD * 2);  // 128KB

    const int tid = threadIdx.x;
    const int wave = tid >> 5, lane = tid & 31;
    const int l15 = lane & 15, lhalf = lane >> 4;
    const int r0 = wave * 16;
    const long long tile = (long long)blockIdx.x * TILE_M;
    const int c0 = lane * 8;

    {
        const unsigned base_lds = lds_lo32(sW);
#pragma unroll 4
        for (int i = tid; i < WBYTES / 16; i += GEMM_THREADS)
            async_g2l_b128(base_lds + i * 16, (const char*)Wp + i * 16);
    }

    for (int rr = 0; rr < 16; ++rr) {
        const int rl = r0 + rr;
        long long rg = tile + rl; if (rg >= rows) rg = rows - 1;
        const float* rp = Tin + rg * DD + c0;
        float4 a0 = *(const float4*)rp;
        float4 a1 = *(const float4*)(rp + 4);
        float vals[8] = {a0.x, a0.y, a0.z, a0.w, a1.x, a1.y, a1.z, a1.w};
        u16x8 o;
#pragma unroll
        for (int i = 0; i < 8; ++i) o[i] = f2bf(vals[i]);
        *(u16x8*)(sA + rl * DD + c0) = o;
    }

    v16bf aF[8];
#pragma unroll
    for (int kt = 0; kt < 8; ++kt)
        aF[kt] = load_a_frag(sA + (r0 + l15) * DD + kt * 32 + lhalf * 8);

    wait_async0();
    __syncthreads();

    for (int nt = 0; nt < 16; ++nt) {
        v8f acc = {0.f, 0.f, 0.f, 0.f, 0.f, 0.f, 0.f, 0.f};
#pragma unroll
        for (int kt = 0; kt < 8; ++kt) {
            v16bf bF = load_b_frag(sW + (((kt * 16 + nt) * 32) + lane) * 16);
            acc = wmma_bf16(aF[kt], bF, acc);
        }
        const int col = nt * 16 + l15;
        const float bv = bias[col];
#pragma unroll
        for (int r = 0; r < 8; ++r) {
            const int rl = r0 + lhalf * 8 + r;
            const long long rg = tile + rl;
            if (rg < rows) Tout[rg * DD + col] = acc[r] + bv;
        }
    }
}

// ---------------------------------------------------------------- logits: logit0 & alpha_r
__global__ __launch_bounds__(256)
void logits_kernel(const float* __restrict__ x1, const float* __restrict__ wl0,
                   const float* __restrict__ wl1, const float* __restrict__ bl0,
                   const float* __restrict__ bl1, float* __restrict__ logit0,
                   float* __restrict__ alpha_r, int n) {
    extern __shared__ float sX[];   // 64 x 256
    const long long base = (long long)blockIdx.x * 64;
    for (int i = threadIdx.x; i < 64 * 256; i += 256) {
        long long r = base + (i >> 8); if (r >= n) r = n - 1;
        sX[i] = x1[r * 256 + (i & 255)];
    }
    __syncthreads();
    const int row = threadIdx.x >> 2;
    const int qd  = threadIdx.x & 3;
    const long long rg = base + row;
    if (rg < n) {
        const float* xr = sX + row * 256;
        for (int hh = 0; hh < 2; ++hh) {
            const int h = qd * 2 + hh;
            const float* w0 = wl0 + h * 256;
            const float* w1 = wl1 + h * 256;
            float d0 = 0.f, d1 = 0.f;
            for (int d = 0; d < 256; ++d) { float xv = xr[d]; d0 += xv * w0[d]; d1 += xv * w1[d]; }
            logit0[rg * 8 + h]  = (d0 + bl0[h]) * INV_SQRT_DQH;
            alpha_r[rg * 8 + h] = d1 + bl1[h];
        }
    }
}

// ---------------------------------------------------------------- order-0 softmax
__global__ __launch_bounds__(256)
void head_max_kernel(const float* __restrict__ logit0, unsigned* __restrict__ hmax,
                     long long n8) {
    __shared__ unsigned bm[8];
    if (threadIdx.x < 8) bm[threadIdx.x] = 0u;
    __syncthreads();
    const long long start  = (long long)blockIdx.x * blockDim.x + threadIdx.x;
    const long long stride = (long long)gridDim.x * blockDim.x;   // multiple of 8
    unsigned m = 0u;
    for (long long i = start; i < n8; i += stride) m = max(m, fenc(logit0[i]));
    atomicMax(&bm[(int)(start & 7)], m);
    __syncthreads();
    if (threadIdx.x < 8) atomicMax(&hmax[threadIdx.x], bm[threadIdx.x]);
}

__global__ __launch_bounds__(256)
void att0_accum_kernel(const float* __restrict__ logit0, const float* __restrict__ v,
                       const unsigned* __restrict__ hmax_enc, float* __restrict__ att0_acc,
                       float* __restrict__ hsum, int n) {
    __shared__ float sacc[256];
    __shared__ float ssum[8];
    sacc[threadIdx.x] = 0.f;
    if (threadIdx.x < 8) ssum[threadIdx.x] = 0.f;
    __syncthreads();
    float hm[8];
#pragma unroll
    for (int h = 0; h < 8; ++h) hm[h] = fdec(hmax_enc[h]);
    float ls[8] = {0.f, 0.f, 0.f, 0.f, 0.f, 0.f, 0.f, 0.f};
    for (long long nd = (long long)blockIdx.x * blockDim.x + threadIdx.x; nd < n;
         nd += (long long)gridDim.x * blockDim.x) {
        float w[8];
#pragma unroll
        for (int h = 0; h < 8; ++h) { w[h] = __expf(logit0[nd * 8 + h] - hm[h]); ls[h] += w[h]; }
        const float* vr = v + nd * 256;
        for (int d = 0; d < 256; ++d) atomicAdd(&sacc[d], w[d >> 5] * vr[d]);
    }
#pragma unroll
    for (int h = 0; h < 8; ++h) atomicAdd(&ssum[h], ls[h]);
    __syncthreads();
    atomicAdd(&att0_acc[threadIdx.x], sacc[threadIdx.x]);
    if (threadIdx.x < 8) atomicAdd(&hsum[threadIdx.x], ssum[threadIdx.x]);
}

// finalize att0 and run its blk2 (single row) in one small kernel
__global__ __launch_bounds__(256)
void att0_blk2_kernel(const float* __restrict__ att0_acc, const float* __restrict__ hsum,
                      const float* __restrict__ ln2g, const float* __restrict__ ln2b,
                      const float* __restrict__ m2w1, const float* __restrict__ b2eff0,
                      const float* __restrict__ m2w2, const float* __restrict__ m2b2,
                      float* __restrict__ att0_out) {
    __shared__ float t[256], lnv[256], hid[256], rs[256], rq[256];
    const int j = threadIdx.x;
    const float tv = att0_acc[j] / hsum[j >> 5];
    t[j] = tv; rs[j] = tv; rq[j] = tv * tv;
    __syncthreads();
    for (int off = 128; off > 0; off >>= 1) {
        if (j < off) { rs[j] += rs[j + off]; rq[j] += rq[j + off]; }
        __syncthreads();
    }
    const float mean = rs[0] * (1.f / 256.f);
    const float var  = rq[0] * (1.f / 256.f) - mean * mean;
    const float rstd = rsqrtf(var + 1e-5f);
    lnv[j] = (tv - mean) * rstd * ln2g[j] + ln2b[j];
    __syncthreads();
    float a = b2eff0[j];
    for (int d = 0; d < 256; ++d) a += lnv[d] * m2w1[d * 256 + j];
    hid[j] = fmaxf(a, 0.f);
    __syncthreads();
    float o = m2b2[j] + t[j];
    for (int d = 0; d < 256; ++d) o += hid[d] * m2w2[d * 256 + j];
    att0_out[j] = o;
}

// ---------------------------------------------------------------- PMA segment softmax
__global__ void seg_max_kernel(const float* __restrict__ alpha_r, const int* __restrict__ node_idx,
                               const int* __restrict__ edge_idx, unsigned* __restrict__ m_enc,
                               long long nnz8) {
    for (long long i = (long long)blockIdx.x * blockDim.x + threadIdx.x; i < nnz8;
         i += (long long)gridDim.x * blockDim.x) {
        const long long e = i >> 3; const int h = (int)(i & 7);
        float a = alpha_r[(long long)edge_idx[e] * 8 + h];
        a = a > 0.f ? a : NEGS * a;
        atomicMax(&m_enc[(long long)node_idx[e] * 8 + h], fenc(a));
    }
}

__global__ void seg_sumw_kernel(const float* __restrict__ alpha_r, const int* __restrict__ node_idx,
                                const int* __restrict__ edge_idx, const unsigned* __restrict__ m_enc,
                                float* __restrict__ s, float* __restrict__ w_edge, long long nnz8) {
    for (long long i = (long long)blockIdx.x * blockDim.x + threadIdx.x; i < nnz8;
         i += (long long)gridDim.x * blockDim.x) {
        const long long e = i >> 3; const int h = (int)(i & 7);
        const long long tg = (long long)node_idx[e] * 8 + h;
        float a = alpha_r[(long long)edge_idx[e] * 8 + h];
        a = a > 0.f ? a : NEGS * a;
        const float w = __expf(a - fdec(m_enc[tg]));
        w_edge[i] = w;
        atomicAdd(&s[tg], w);
    }
}

__global__ void norm_w_kernel(float* __restrict__ w_edge, const int* __restrict__ node_idx,
                              const float* __restrict__ s, long long nnz8) {
    for (long long i = (long long)blockIdx.x * blockDim.x + threadIdx.x; i < nnz8;
         i += (long long)gridDim.x * blockDim.x) {
        const long long e = i >> 3; const int h = (int)(i & 7);
        w_edge[i] = w_edge[i] / (s[(long long)node_idx[e] * 8 + h] + 1e-16f);
    }
}

__global__ void msg_scatter_kernel(const float* __restrict__ w_edge, const int* __restrict__ node_idx,
                                   const int* __restrict__ edge_idx, const float* __restrict__ v,
                                   float* __restrict__ xe, long long total) {
    for (long long i = (long long)blockIdx.x * blockDim.x + threadIdx.x; i < total;
         i += (long long)gridDim.x * blockDim.x) {
        const long long e = i >> 8; const int d = (int)(i & 255);
        const int tgt = node_idx[e];
        if (tgt < EE) {
            const float wv = w_edge[e * 8 + (d >> 5)];
            atomicAdd(&xe[(long long)tgt * 256 + d], v[(long long)edge_idx[e] * 256 + d] * wv);
        }
    }
}

// ---------------------------------------------------------------- host launcher
extern "C" void kernel_launch(void* const* d_in, const int* in_sizes, int n_in,
                              void* d_out, int out_size, void* d_ws, size_t ws_size,
                              hipStream_t stream) {
    const float* x        = (const float*)d_in[0];
    const int* node_idx   = (const int*)d_in[1];
    const int* edge_idx   = (const int*)d_in[2];
    const int* edge_ord   = (const int*)d_in[3];
    const float* pe_q = (const float*)d_in[4];
    const float* qw1  = (const float*)d_in[5];
    const float* qb1  = (const float*)d_in[6];
    const float* qw2  = (const float*)d_in[7];
    const float* qb2  = (const float*)d_in[8];
    const float* kw   = (const float*)d_in[9];
    const float* kb   = (const float*)d_in[10];
    const float* vw   = (const float*)d_in[11];
    const float* vb   = (const float*)d_in[12];
    const float* m1w1 = (const float*)d_in[13];
    const float* m1b1 = (const float*)d_in[14];
    const float* m1w2 = (const float*)d_in[15];
    const float* m1b2 = (const float*)d_in[16];
    const float* m2w1 = (const float*)d_in[17];
    const float* m2b1 = (const float*)d_in[18];
    const float* m2w2 = (const float*)d_in[19];
    const float* m2b2 = (const float*)d_in[20];
    const float* m3w1 = (const float*)d_in[21];
    const float* m3b1 = (const float*)d_in[22];
    const float* m3w2 = (const float*)d_in[23];
    const float* m3b2 = (const float*)d_in[24];
    const float* ln1g = (const float*)d_in[25];
    const float* ln1b = (const float*)d_in[26];
    const float* ln2g = (const float*)d_in[27];
    const float* ln2b = (const float*)d_in[28];
    const float* ln3g = (const float*)d_in[29];
    const float* ln3b = (const float*)d_in[30];
    const float* pe2  = (const float*)d_in[31];
    const float* pe3  = (const float*)d_in[32];
    const float* pe_b = (const float*)d_in[33];
    const float* bw1  = (const float*)d_in[34];
    const float* bb1  = (const float*)d_in[35];
    const float* bw2  = (const float*)d_in[36];
    const float* bb2  = (const float*)d_in[37];

    // -------- workspace carve-up
    char* base = (char*)d_ws;
    size_t off = 0;
    auto alloc = [&](size_t bytes) -> char* {
        char* p = base + off;
        off = (off + bytes + 255) & ~(size_t)255;
        return p;
    };
    float* wl0      = (float*)alloc(8 * 256 * 4);
    float* wl1      = (float*)alloc(8 * 256 * 4);
    float* bl0      = (float*)alloc(8 * 4);
    float* bl1      = (float*)alloc(8 * 4);
    float* bias_out = (float*)alloc(MAXL1 * 256 * 4);
    float* b2eff    = (float*)alloc(2 * 256 * 4);
    float* c3       = (float*)alloc(MAXL1 * 256 * 4);
    float* att0_acc = (float*)alloc(256 * 4);
    float* hsum     = (float*)alloc(8 * 4);
    unsigned* hmax  = (unsigned*)alloc(8 * 4);
    float* att0_b2  = (float*)alloc(256 * 4);
    unsigned short* m1w1p = (unsigned short*)alloc(65536 * 2);
    unsigned short* m1w2p = (unsigned short*)alloc(65536 * 2);
    unsigned short* m2w1p = (unsigned short*)alloc(65536 * 2);
    unsigned short* m2w2p = (unsigned short*)alloc(65536 * 2);
    unsigned short* m3w1p = (unsigned short*)alloc(65536 * 2);
    unsigned short* m3w2p = (unsigned short*)alloc(65536 * 2);
    unsigned short* vwp   = (unsigned short*)alloc(65536 * 2);
    float* x1     = (float*)alloc((size_t)NN * 256 * 4);   // x after MLP1; reused as x_v
    float* vbuf   = (float*)alloc((size_t)NN * 256 * 4);
    float* xe     = (float*)alloc((size_t)EE * 256 * 4);
    float* logit0 = (float*)alloc((size_t)NN * 8 * 4);
    float* aralph = (float*)alloc((size_t)NN * 8 * 4);
    unsigned* menc = (unsigned*)alloc((size_t)NN * 8 * 4);
    float* sseg   = (float*)alloc((size_t)NN * 8 * 4);
    float* w_edge = (float*)alloc((size_t)NNZ * 8 * 4);

    // -------- zero accumulators (fenc ordering: 0 <= fenc(-inf), so 0-init is valid)
    hipMemsetAsync(att0_acc, 0, 256 * 4, stream);
    hipMemsetAsync(hsum, 0, 8 * 4, stream);
    hipMemsetAsync(hmax, 0, 8 * 4, stream);
    hipMemsetAsync(menc, 0, (size_t)NN * 8 * 4, stream);
    hipMemsetAsync(sseg, 0, (size_t)NN * 8 * 4, stream);
    hipMemsetAsync(xe, 0, (size_t)EE * 256 * 4, stream);

    // -------- prep: tiny MLPs + algebraic folds
    small_prep_kernel<<<1, 256, 0, stream>>>(pe_q, qw1, qb1, qw2, qb2, kw, kb,
                                             pe_b, bw1, bb1, bw2, bb2,
                                             pe2, m2w1, m2b1, pe3, m3w1, m3b1,
                                             wl0, wl1, bl0, bl1, bias_out, b2eff, c3);
    pack_weight_kernel<<<256, 256, 0, stream>>>(m1w1, m1w1p);
    pack_weight_kernel<<<256, 256, 0, stream>>>(m1w2, m1w2p);
    pack_weight_kernel<<<256, 256, 0, stream>>>(m2w1, m2w1p);   // top 256 rows of [512,256]
    pack_weight_kernel<<<256, 256, 0, stream>>>(m2w2, m2w2p);
    pack_weight_kernel<<<256, 256, 0, stream>>>(m3w1, m3w1p);   // top 256 rows
    pack_weight_kernel<<<256, 256, 0, stream>>>(m3w2, m3w2p);
    pack_weight_kernel<<<256, 256, 0, stream>>>(vw, vwp);

    const int gN = (NN + TILE_M - 1) / TILE_M;
    const int gE = (EE + TILE_M - 1) / TILE_M;
    const size_t smF = (size_t)TILE_M * DD * 2 * 2 + WBYTES;   // 32K+32K+128K = 192KB
    const size_t smG = (size_t)TILE_M * DD * 2 + WBYTES;       // 32K+128K   = 160KB

    // x1 = x + MLP1(LN1(x))
    fused_block_kernel<<<gN, GEMM_THREADS, smF, stream>>>(
        x, x1, NN, ln1g, ln1b, m1w1p, m1b1, nullptr, 0, m1w2p, m1b2,
        nullptr, nullptr, 0);

    // v = x1 @ vw + vb
    gemm_bias_kernel<<<gN, GEMM_THREADS, smG, stream>>>(x1, vbuf, NN, vwp, vb);

    // logit0 (scaled) and alpha_r via folded kw*q
    logits_kernel<<<(NN + 63) / 64, 256, 64 * 256 * 4, stream>>>(
        x1, wl0, wl1, bl0, bl1, logit0, aralph, NN);

    // order-0 global softmax + weighted sum of v
    head_max_kernel<<<512, 256, 0, stream>>>(logit0, hmax, (long long)NN * 8);
    att0_accum_kernel<<<256, 256, 0, stream>>>(logit0, vbuf, hmax, att0_acc, hsum, NN);
    att0_blk2_kernel<<<1, 256, 0, stream>>>(att0_acc, hsum, ln2g, ln2b,
                                            m2w1, b2eff /* type 0 */, m2w2, m2b2, att0_b2);

    // PMA segment softmax + scatter
    seg_max_kernel<<<2048, 256, 0, stream>>>(aralph, node_idx, edge_idx, menc,
                                             (long long)NNZ * 8);
    seg_sumw_kernel<<<2048, 256, 0, stream>>>(aralph, node_idx, edge_idx, menc,
                                              sseg, w_edge, (long long)NNZ * 8);
    norm_w_kernel<<<2048, 256, 0, stream>>>(w_edge, node_idx, sseg, (long long)NNZ * 8);
    msg_scatter_kernel<<<4096, 256, 0, stream>>>(w_edge, node_idx, edge_idx, vbuf, xe,
                                                 (long long)NNZ * 256);

    // blk2 on att1_v (= v) -> x_v (reuses x1 buffer), adding att0_blk2 broadcast
    fused_block_kernel<<<gN, GEMM_THREADS, smF, stream>>>(
        vbuf, x1, NN, ln2g, ln2b, m2w1p, b2eff, nullptr, 1, m2w2p, m2b2,
        att0_b2, nullptr, 0);
    // blk2 on att1_e (in place), adding att0_blk2
    fused_block_kernel<<<gE, GEMM_THREADS, smF, stream>>>(
        xe, xe, EE, ln2g, ln2b, m2w1p, b2eff, nullptr, 1, m2w2p, m2b2,
        att0_b2, nullptr, 0);

    // blk3 + final bias, writing straight to d_out
    float* out_v = (float*)d_out;
    float* out_e = (float*)d_out + (size_t)NN * 256;
    fused_block_kernel<<<gN, GEMM_THREADS, smF, stream>>>(
        x1, out_v, NN, ln3g, ln3b, m3w1p, c3, nullptr, 1, m3w2p, m3b2,
        nullptr, bias_out, 1);
    fused_block_kernel<<<gE, GEMM_THREADS, smF, stream>>>(
        xe, out_e, EE, ln3g, ln3b, m3w1p, c3, edge_ord, 0, m3w2p, m3b2,
        nullptr, bias_out, 0);
}
